// MAMFGCN_85822036509064
// MI455X (gfx1250) — matrix operand order repaired and checked by reading, not compiled, not run on hardware
//
#include <hip/hip_runtime.h>
#include <hip/hip_bf16.h>
#include <stdint.h>

// ---------------- problem constants ----------------
#define GN    8192   // nodes
#define NFEAT 512
#define NHID  256
#define NOUT  128
#define INPW  1536   // NFEAT + 4*NHID (concat width)

// ---------------- GEMM tiling ----------------
#define BM 64
#define BN 128
#define BK 32
#define LSTR 34      // padded LDS stride, f32-convert path (bf16 elems)
#define ASTR 40      // padded LDS stride, async path (80B rows: 16B aligned)

typedef __attribute__((ext_vector_type(16))) __bf16       v16bf;
typedef __attribute__((ext_vector_type(8)))  float        v8f;
typedef __attribute__((ext_vector_type(8)))  unsigned int v8u;

union Frag { v8u u; v16bf b; };

__device__ __forceinline__ unsigned int pack_bf16(float a, float b) {
  union { __bf16 h[2]; unsigned int u; } p;
  p.h[0] = (__bf16)a;
  p.h[1] = (__bf16)b;
  return p.u;
}
__device__ __forceinline__ unsigned short cvt1_bf16(float a) {
  union { __bf16 h; unsigned short u; } p;
  p.h = (__bf16)a;
  return p.u;
}

__device__ __forceinline__ float breduce(float v, float* red, int c, int n) {
  red[c] = v; __syncthreads();
  for (int s = n >> 1; s > 0; s >>= 1) {
    if (c < s) red[c] += red[c + s];
    __syncthreads();
  }
  float r = red[0];
  __syncthreads();
  return r;
}

// ---------------------------------------------------------------------------
// Variant 1: A,B f32 row-major, converted to bf16 on the fly in LDS.
// Used for the skinny inp@W GEMMs and as the small-workspace fallback.
// ---------------------------------------------------------------------------
__global__ __launch_bounds__(256) void gemm_bf16_wmma(
    const float* __restrict__ A, int lda,
    const float* __restrict__ B, int ldb,
    float* __restrict__ C, int ldc,
    const float* __restrict__ bias, int K)
{
  __shared__ unsigned short As[BM * LSTR];   // [m][k]
  __shared__ unsigned short Bt[BN * LSTR];   // [n][k] (transposed)

  const int tid  = threadIdx.x;
  const int lane = tid & 31;
  const int wv   = tid >> 5;
  const int wm   = wv & 1;
  const int wn   = wv >> 1;
  const int l15  = lane & 15;
  const int lh   = lane >> 4;
  const int rowBase = blockIdx.y * BM;
  const int colBase = blockIdx.x * BN;

  v8f zero = {};
  v8f acc[2][2];
  acc[0][0] = zero; acc[0][1] = zero; acc[1][0] = zero; acc[1][1] = zero;

  for (int k0 = 0; k0 < K; k0 += BK) {
#pragma unroll
    for (int it = 0; it < 2; ++it) {          // A tile 64x32
      int idx = tid + it * 256;
      int r   = idx >> 3;
      int c4  = (idx & 7) << 2;
      float4 v = *(const float4*)(A + (size_t)(rowBase + r) * lda + (k0 + c4));
      unsigned int* d = (unsigned int*)&As[r * LSTR + c4];
      d[0] = pack_bf16(v.x, v.y);
      d[1] = pack_bf16(v.z, v.w);
    }
#pragma unroll
    for (int it = 0; it < 4; ++it) {          // B tile 32x128 -> transposed
      int idx = tid + it * 256;
      int kk  = idx >> 5;
      int n4  = (idx & 31) << 2;
      float4 v = *(const float4*)(B + (size_t)(k0 + kk) * ldb + (colBase + n4));
      Bt[(n4 + 0) * LSTR + kk] = cvt1_bf16(v.x);
      Bt[(n4 + 1) * LSTR + kk] = cvt1_bf16(v.y);
      Bt[(n4 + 2) * LSTR + kk] = cvt1_bf16(v.z);
      Bt[(n4 + 3) * LSTR + kk] = cvt1_bf16(v.w);
    }
    if (k0 + BK < K) {
      if (tid < BM)
        __builtin_prefetch(A + (size_t)(rowBase + tid) * lda + (k0 + BK), 0, 0);
      else if (tid < BM + BK)
        __builtin_prefetch(B + (size_t)(k0 + BK + (tid - BM)) * ldb + colBase, 0, 0);
    }
    __syncthreads();

    Frag afr[2], bfr[2];
#pragma unroll
    for (int i = 0; i < 2; ++i) {
      int m = wm * 32 + i * 16 + l15;
#pragma unroll
      for (int v = 0; v < 8; ++v) {
        int kk = (v >> 2) * 16 + lh * 8 + (v & 3) * 2;
        afr[i].u[v] = *(const unsigned int*)&As[m * LSTR + kk];
      }
    }
#pragma unroll
    for (int j = 0; j < 2; ++j) {
      int n = wn * 32 + j * 16 + l15;
#pragma unroll
      for (int v = 0; v < 8; ++v) {
        int kk = (v >> 2) * 16 + lh * 8 + (v & 3) * 2;
        bfr[j].u[v] = *(const unsigned int*)&Bt[n * LSTR + kk];
      }
    }
#pragma unroll
    for (int i = 0; i < 2; ++i)
#pragma unroll
      for (int j = 0; j < 2; ++j)
        acc[i][j] = __builtin_amdgcn_wmma_f32_16x16x32_bf16(
            false, afr[i].b, false, bfr[j].b, (short)0, acc[i][j], false, false);
    __syncthreads();
  }

#pragma unroll
  for (int i = 0; i < 2; ++i)
#pragma unroll
    for (int j = 0; j < 2; ++j) {
      int n = colBase + wn * 32 + j * 16 + l15;
      float bv = bias ? bias[n] : 0.0f;
#pragma unroll
      for (int v = 0; v < 8; ++v) {
        int m = rowBase + wm * 32 + i * 16 + lh * 8 + v;
        C[(size_t)m * ldc + n] = acc[i][j][v] + bv;
      }
    }
}

// ---------------------------------------------------------------------------
// Variant 2: A (MxK) and Bt (NxK) both bf16 in global memory.  Both tiles
// streamed with global_load_async_to_lds_b128, double-buffered in LDS with
// counted s_wait_asynccnt so DMA of tile t+1 overlaps WMMAs of tile t.
// ---------------------------------------------------------------------------
__global__ __launch_bounds__(256) void gemm_async_bf16_wmma(
    const unsigned short* __restrict__ Abf, int lda,    // M x K, bf16
    const unsigned short* __restrict__ BTbf, int ldb,   // N x K, bf16
    float* __restrict__ C, int ldc,
    const float* __restrict__ bias, int K)
{
  __shared__ unsigned short As[2][BM * ASTR];
  __shared__ unsigned short Bs[2][BN * ASTR];

  const int tid  = threadIdx.x;
  const int lane = tid & 31;
  const int wv   = tid >> 5;
  const int wm   = wv & 1;
  const int wn   = wv >> 1;
  const int l15  = lane & 15;
  const int lh   = lane >> 4;
  const int rowBase = blockIdx.y * BM;
  const int colBase = blockIdx.x * BN;

  // A: 64x32 bf16 tile = 256 x 16B chunks, one per thread
  const int ra = tid >> 2, sa = tid & 3;
  const unsigned a_gbase = ((unsigned)(rowBase + ra) * (unsigned)lda + (unsigned)(sa * 8)) * 2u;
  const unsigned a_lds[2] = {
      (unsigned)(size_t)&As[0][ra * ASTR + sa * 8],
      (unsigned)(size_t)&As[1][ra * ASTR + sa * 8] };
  // B: 128x32 bf16 tile = 512 x 16B chunks, two per thread
  const int id0 = tid * 2, id1 = tid * 2 + 1;
  const int rb0 = id0 >> 2, sb0 = id0 & 3;
  const int rb1 = id1 >> 2, sb1 = id1 & 3;
  const unsigned b_gbase0 = ((unsigned)(colBase + rb0) * (unsigned)ldb + (unsigned)(sb0 * 8)) * 2u;
  const unsigned b_gbase1 = ((unsigned)(colBase + rb1) * (unsigned)ldb + (unsigned)(sb1 * 8)) * 2u;
  const unsigned b_lds0[2] = {
      (unsigned)(size_t)&Bs[0][rb0 * ASTR + sb0 * 8],
      (unsigned)(size_t)&Bs[1][rb0 * ASTR + sb0 * 8] };
  const unsigned b_lds1[2] = {
      (unsigned)(size_t)&Bs[0][rb1 * ASTR + sb1 * 8],
      (unsigned)(size_t)&Bs[1][rb1 * ASTR + sb1 * 8] };

  auto issue = [&](int buf, int k0) {
    unsigned ko = (unsigned)k0 * 2u;
    asm volatile("global_load_async_to_lds_b128 %0, %1, %2"
                 :: "v"(a_lds[buf]), "v"(a_gbase + ko), "s"(Abf) : "memory");
    asm volatile("global_load_async_to_lds_b128 %0, %1, %2"
                 :: "v"(b_lds0[buf]), "v"(b_gbase0 + ko), "s"(BTbf) : "memory");
    asm volatile("global_load_async_to_lds_b128 %0, %1, %2"
                 :: "v"(b_lds1[buf]), "v"(b_gbase1 + ko), "s"(BTbf) : "memory");
  };

  v8f zero = {};
  v8f acc[2][2];
  acc[0][0] = zero; acc[0][1] = zero; acc[1][0] = zero; acc[1][1] = zero;

  const int nk = K / BK;
  issue(0, 0);                                  // prologue: tile 0
  for (int t = 0; t < nk; ++t) {
    const int cur = t & 1;
    if (t + 1 < nk) {
      issue(cur ^ 1, (t + 1) * BK);             // stream next tile behind compute
      asm volatile("s_wait_asynccnt 0x3" ::: "memory");  // tile t done (newest 3 = t+1)
    } else {
      asm volatile("s_wait_asynccnt 0x0" ::: "memory");
    }
    __syncthreads();

    const unsigned short* Ab = As[cur];
    const unsigned short* Bb = Bs[cur];
    Frag afr[2], bfr[2];
#pragma unroll
    for (int i = 0; i < 2; ++i) {
      int m = wm * 32 + i * 16 + l15;
#pragma unroll
      for (int v = 0; v < 8; ++v) {
        int kk = (v >> 2) * 16 + lh * 8 + (v & 3) * 2;
        afr[i].u[v] = *(const unsigned int*)&Ab[m * ASTR + kk];
      }
    }
#pragma unroll
    for (int j = 0; j < 2; ++j) {
      int n = wn * 32 + j * 16 + l15;
#pragma unroll
      for (int v = 0; v < 8; ++v) {
        int kk = (v >> 2) * 16 + lh * 8 + (v & 3) * 2;
        bfr[j].u[v] = *(const unsigned int*)&Bb[n * ASTR + kk];
      }
    }
#pragma unroll
    for (int i = 0; i < 2; ++i)
#pragma unroll
      for (int j = 0; j < 2; ++j)
        acc[i][j] = __builtin_amdgcn_wmma_f32_16x16x32_bf16(
            false, afr[i].b, false, bfr[j].b, (short)0, acc[i][j], false, false);
    __syncthreads();
  }

#pragma unroll
  for (int i = 0; i < 2; ++i)
#pragma unroll
    for (int j = 0; j < 2; ++j) {
      int n = colBase + wn * 32 + j * 16 + l15;
      float bv = bias ? bias[n] : 0.0f;
#pragma unroll
      for (int v = 0; v < 8; ++v) {
        int m = rowBase + wm * 32 + i * 16 + lh * 8 + v;
        C[(size_t)m * ldc + n] = acc[i][j][v] + bv;
      }
    }
}

// ---------------------------------------------------------------------------
// helpers
// ---------------------------------------------------------------------------
__global__ void cvt_f32_bf16_kernel(const float* __restrict__ src,
                                    unsigned short* __restrict__ dst) {
  size_t idx = ((size_t)blockIdx.x * 256 + threadIdx.x) * 4;
  float4 v = *(const float4*)(src + idx);
  uint2 p;
  p.x = pack_bf16(v.x, v.y);
  p.y = pack_bf16(v.z, v.w);
  *(uint2*)(dst + idx) = p;
}

// tmp (GN x Nc, f32) -> tmpT (Nc x GN, bf16); grid (GN/32, Nc/32), 256 thr
__global__ __launch_bounds__(256) void transpose_cvt_kernel(
    const float* __restrict__ src, unsigned short* __restrict__ dst, int Nc) {
  __shared__ unsigned short t[32][34];
  int bx = blockIdx.x * 32;               // GN dim
  int by = blockIdx.y * 32;               // Nc dim
  int tx = threadIdx.x & 31, ty = threadIdx.x >> 5;
#pragma unroll
  for (int i = 0; i < 32; i += 8)
    t[ty + i][tx] = cvt1_bf16(src[(size_t)(bx + ty + i) * Nc + (by + tx)]);
  __syncthreads();
#pragma unroll
  for (int i = 0; i < 32; i += 8)
    dst[(size_t)(by + ty + i) * GN + (bx + tx)] = t[tx][ty + i];
}

__global__ void copy_x_kernel(const float* __restrict__ x, float* __restrict__ inp) {
  int idx = blockIdx.x * 256 + threadIdx.x;  // GN*NFEAT
  int r = idx >> 9, c = idx & 511;
  inp[(size_t)r * INPW + c] = x[idx];
}

__global__ __launch_bounds__(256) void col_stats_kernel(
    const float* __restrict__ z, float* __restrict__ part, float* __restrict__ sqp) {
  int c = threadIdx.x;
  int b = blockIdx.x;
  int r0 = b * 64;
  float s = 0.f, q = 0.f;
  for (int r = 0; r < 64; ++r) {
    float v = z[(size_t)(r0 + r) * NHID + c];
    s += v; q += v * v;
  }
  part[b * NHID + c] = s;
  __shared__ float red[256];
  float tot = breduce(q, red, c, 256);
  if (c == 0) sqp[b] = tot;
}

__global__ __launch_bounds__(256) void finalize_stats_kernel(
    const float* __restrict__ part, const float* __restrict__ sqp,
    float* __restrict__ colmean, float* __restrict__ scale) {
  int c = threadIdx.x;
  float s = 0.f;
  for (int b = 0; b < 128; ++b) s += part[b * NHID + c];
  colmean[c] = s * (1.0f / (float)GN);
  __shared__ float red[256];
  float q = (c < 128) ? sqp[c] : 0.0f;
  float total = breduce(q, red, c, 256);
  float s2    = breduce(s * s, red, c, 256);
  if (c == 0) {
    float centered = total - s2 * (1.0f / (float)GN);
    scale[0] = 1.0f / sqrtf(1e-6f + centered * (1.0f / (float)GN));
  }
}

__global__ void pairnorm_tanh_kernel(const float* __restrict__ z,
    const float* __restrict__ colmean, const float* __restrict__ scale,
    float* __restrict__ dst) {
  int idx = blockIdx.x * 256 + threadIdx.x;
  int r = idx >> 8, c = idx & 255;
  float v = (z[idx] - colmean[c]) * scale[0];
  dst[(size_t)r * INPW + c] = tanhf(v);
}

__global__ __launch_bounds__(128) void l2norm_kernel(
    const float* __restrict__ z, float* __restrict__ out) {
  int r = blockIdx.x, c = threadIdx.x;
  __shared__ float red[128];
  float v = z[(size_t)r * NOUT + c];
  float s = breduce(v * v, red, c, 128);
  float nrm = fmaxf(sqrtf(s), 1e-12f);
  out[(size_t)r * NOUT + c] = v / nrm;
}

__global__ __launch_bounds__(128) void attention_kernel(
    const float* __restrict__ emb1, const float* __restrict__ emb2,
    const float* __restrict__ emb3, const float* __restrict__ com1,
    const float* __restrict__ com2, const float* __restrict__ com3,
    const float* __restrict__ W1, const float* __restrict__ b1,
    const float* __restrict__ W2, const float* __restrict__ Wm,
    const float* __restrict__ bm,
    float* __restrict__ out_prob, float* __restrict__ out_beta) {
  int r = blockIdx.x, c = threadIdx.x;
  __shared__ float red[128];
  size_t off = (size_t)r * NOUT + c;
  float zv[4];
  zv[0] = emb1[off];
  zv[1] = emb2[off];
  zv[2] = emb3[off];
  zv[3] = (com1[off] + com2[off] + com3[off]) * (1.0f / 3.0f);
  float w10 = W1[c * 2 + 0], w11 = W1[c * 2 + 1];
  float wv[4];
  for (int v = 0; v < 4; ++v) {
    float d0 = breduce(zv[v] * w10, red, c, 128);
    float d1 = breduce(zv[v] * w11, red, c, 128);
    float h0 = tanhf(d0 + b1[0]);
    float h1 = tanhf(d1 + b1[1]);
    wv[v] = h0 * W2[0] + h1 * W2[1];
  }
  float mx = fmaxf(fmaxf(wv[0], wv[1]), fmaxf(wv[2], wv[3]));
  float e[4], se = 0.f;
  for (int v = 0; v < 4; ++v) { e[v] = expf(wv[v] - mx); se += e[v]; }
  float beta[4];
  for (int v = 0; v < 4; ++v) beta[v] = e[v] / se;
  if (c < 4) out_beta[(size_t)r * 4 + c] = beta[c];
  float emb = 0.f;
  for (int v = 0; v < 4; ++v) emb += beta[v] * zv[v];
  float l0 = breduce(emb * Wm[c * 2 + 0], red, c, 128);
  float l1 = breduce(emb * Wm[c * 2 + 1], red, c, 128);
  if (c == 0) {
    l0 += bm[0]; l1 += bm[1];
    float m2 = fmaxf(l0, l1);
    float e0 = expf(l0 - m2), e1 = expf(l1 - m2);
    float inv = 1.0f / (e0 + e1);
    out_prob[(size_t)r * 2 + 0] = e0 * inv;
    out_prob[(size_t)r * 2 + 1] = e1 * inv;
  }
}

// ---------------------------------------------------------------------------
static void launch_gemm(hipStream_t s, const float* A, int lda, const float* B,
                        int ldb, float* C, int ldc, const float* bias,
                        int M, int Ncols, int K) {
  dim3 grid(Ncols / BN, M / BM);
  gemm_bf16_wmma<<<grid, 256, 0, s>>>(A, lda, B, ldb, C, ldc, bias, K);
}

static void launch_gemm_async(hipStream_t s, const unsigned short* A, int lda,
                              const unsigned short* BT, int ldb, float* C,
                              int ldc, const float* bias, int M, int Ncols, int K) {
  dim3 grid(Ncols / BN, M / BM);
  gemm_async_bf16_wmma<<<grid, 256, 0, s>>>(A, lda, BT, ldb, C, ldc, bias, K);
}

extern "C" void kernel_launch(void* const* d_in, const int* in_sizes, int n_in,
                              void* d_out, int out_size, void* d_ws, size_t ws_size,
                              hipStream_t stream) {
  (void)in_sizes; (void)n_in; (void)out_size;

  const float* x0    = (const float*)d_in[0];
  const float* x1    = (const float*)d_in[1];
  const float* sadj  = (const float*)d_in[2];
  const float* fadj  = (const float*)d_in[3];
  const float* fadj2 = (const float*)d_in[4];
  void* const* P1 = d_in + 5;
  void* const* P2 = d_in + 15;
  void* const* P3 = d_in + 25;
  void* const* PC = d_in + 35;
  const float* attW1 = (const float*)d_in[45];
  const float* attb1 = (const float*)d_in[46];
  const float* attW2 = (const float*)d_in[47];
  const float* mlpW  = (const float*)d_in[48];
  const float* mlpb  = (const float*)d_in[49];

  float* out    = (float*)d_out;
  float* o_out  = out;
  float* o_beta = out + (size_t)GN * 2;
  float* o_emb1 = o_beta + (size_t)GN * 4;
  float* o_com1 = o_emb1 + (size_t)GN * NOUT;
  float* o_com2 = o_com1 + (size_t)GN * NOUT;
  float* o_com3 = o_com2 + (size_t)GN * NOUT;
  float* o_emb2 = o_com3 + (size_t)GN * NOUT;
  float* o_emb3 = o_emb2 + (size_t)GN * NOUT;

  float* ws      = (float*)d_ws;
  float* w_inp   = ws;                               // GN x 1536
  float* w_tmp   = w_inp + (size_t)GN * INPW;        // GN x 256
  float* w_z     = w_tmp + (size_t)GN * NHID;        // GN x 256
  float* w_part  = w_z + (size_t)GN * NHID;          // 128 x 256
  float* w_sqp   = w_part + 128 * NHID;              // 128
  float* w_cm    = w_sqp + 128;                      // 256
  float* w_scale = w_cm + NHID;                      // 1

  // bf16 regions (aligned): tmpT (Nc x GN) + 3 adjacency mirrors
  size_t stats_end = (size_t)((char*)(w_scale + 1) - (char*)ws);
  size_t tmpT_off  = (stats_end + 255) & ~(size_t)255;
  size_t adj_elems = (size_t)GN * GN;
  size_t bf_off    = (tmpT_off + (size_t)NHID * GN * 2 + 255) & ~(size_t)255;
  bool   use_bf    = ws_size >= bf_off + 3 * adj_elems * sizeof(unsigned short);

  unsigned short* w_tmpT = (unsigned short*)((char*)d_ws + tmpT_off);
  unsigned short* adjbf[3] = {nullptr, nullptr, nullptr};
  if (use_bf) {
    adjbf[0] = (unsigned short*)((char*)d_ws + bf_off);
    adjbf[1] = adjbf[0] + adj_elems;
    adjbf[2] = adjbf[1] + adj_elems;
    int blocks = (int)(adj_elems / 1024);
    cvt_f32_bf16_kernel<<<blocks, 256, 0, stream>>>(sadj,  adjbf[0]);
    cvt_f32_bf16_kernel<<<blocks, 256, 0, stream>>>(fadj,  adjbf[1]);
    cvt_f32_bf16_kernel<<<blocks, 256, 0, stream>>>(fadj2, adjbf[2]);
  }

  auto run_snowball = [&](const float* x, const float* adj,
                          const unsigned short* abf, void* const* P,
                          float* emb_out) {
    copy_x_kernel<<<GN * NFEAT / 256, 256, 0, stream>>>(x, w_inp);
    for (int k = 0; k < 4; ++k) {
      int fin = NFEAT + k * NHID;
      const float* W = (const float*)P[2 * k];
      const float* b = (const float*)P[2 * k + 1];
      launch_gemm(stream, w_inp, INPW, W, NHID, w_tmp, NHID, nullptr, GN, NHID, fin);
      if (abf) {
        dim3 tg(GN / 32, NHID / 32);
        transpose_cvt_kernel<<<tg, 256, 0, stream>>>(w_tmp, w_tmpT, NHID);
        launch_gemm_async(stream, abf, GN, w_tmpT, GN, w_z, NHID, b, GN, NHID, GN);
      } else {
        launch_gemm(stream, adj, GN, w_tmp, NHID, w_z, NHID, b, GN, NHID, GN);
      }
      col_stats_kernel<<<128, 256, 0, stream>>>(w_z, w_part, w_sqp);
      finalize_stats_kernel<<<1, 256, 0, stream>>>(w_part, w_sqp, w_cm, w_scale);
      pairnorm_tanh_kernel<<<GN * NHID / 256, 256, 0, stream>>>(
          w_z, w_cm, w_scale, w_inp + NFEAT + k * NHID);
    }
    const float* Wo = (const float*)P[8];
    const float* bo = (const float*)P[9];
    launch_gemm(stream, w_inp, INPW, Wo, NOUT, w_tmp, NOUT, nullptr, GN, NOUT, INPW);
    if (abf) {
      dim3 tg(GN / 32, NOUT / 32);
      transpose_cvt_kernel<<<tg, 256, 0, stream>>>(w_tmp, w_tmpT, NOUT);
      launch_gemm_async(stream, abf, GN, w_tmpT, GN, w_z, NOUT, bo, GN, NOUT, GN);
    } else {
      launch_gemm(stream, adj, GN, w_tmp, NOUT, w_z, NOUT, bo, GN, NOUT, GN);
    }
    l2norm_kernel<<<GN, 128, 0, stream>>>(w_z, emb_out);
  };

  run_snowball(x1, sadj,  adjbf[0], P1, o_emb1);
  run_snowball(x0, fadj,  adjbf[1], P2, o_emb2);
  run_snowball(x0, fadj2, adjbf[2], P3, o_emb3);
  run_snowball(x1, sadj,  adjbf[0], PC, o_com1);
  run_snowball(x0, fadj,  adjbf[1], PC, o_com2);
  run_snowball(x0, fadj2, adjbf[2], PC, o_com3);

  attention_kernel<<<GN, 128, 0, stream>>>(o_emb1, o_emb2, o_emb3,
      o_com1, o_com2, o_com3, attW1, attb1, attW2, mlpW, mlpb, o_out, o_beta);
}
